// PhonSSM_84516366451054
// MI455X (gfx1250) — matrix operand order repaired and checked by reading, not compiled, not run on hardware
//
#include <hip/hip_runtime.h>
#include <hip/hip_bf16.h>
#include <math.h>

// ---------------------------------------------------------------------------
// CDNA5 (gfx1250) implementation. wave32, WMMA f32_16x16x32_f16 for GEMMs.
// Weights pre-packed into B-fragment-major layout: one v16h (32B) load per
// lane per 16x16x32 WMMA step.
// ---------------------------------------------------------------------------

typedef __attribute__((ext_vector_type(16))) _Float16 v16h;
typedef __attribute__((ext_vector_type(8)))  float    v8f;

#define BB      64
#define TT      256
#define NNODE   21
#define HIDD    128
#define DOUTD   256
#define DMODELD 256
#define DINNERD 512
#define DSTATED 16
#define NBT     (BB*TT)          // 16384
#define TEMP_INV (1.0f/0.07f)

// ---------------- WMMA fragment helpers (per ISA 7.12.2 layouts) -----------
// A 16x32 f16: lanes 0-15: row=m0+lane,    K = {0..7} e0-7, {16..23} e8-15
//              lanes16-31: row=m0+lane-16, K = {8..15} e0-7, {24..31} e8-15
// (i.e. lane's Ks = k0 + half*8 + {0..7} and k0 + half*8 + {16..23})

// Fast A loader: no guards, M%16==0, K%32==0, lda%4==0. Four float4 loads.
__device__ __forceinline__ v16h frag_a_fast(const float* __restrict__ A, int lda,
                                            int m0, int k0, int lane) {
  const float* p = A + (long)(m0 + (lane & 15)) * lda + k0 + (lane >> 4) * 8;
  float4 a0 = *(const float4*)(p);
  float4 a1 = *(const float4*)(p + 4);
  float4 b0 = *(const float4*)(p + 16);
  float4 b1 = *(const float4*)(p + 20);
  v16h a;
  a[0]  = (_Float16)a0.x; a[1]  = (_Float16)a0.y; a[2]  = (_Float16)a0.z; a[3]  = (_Float16)a0.w;
  a[4]  = (_Float16)a1.x; a[5]  = (_Float16)a1.y; a[6]  = (_Float16)a1.z; a[7]  = (_Float16)a1.w;
  a[8]  = (_Float16)b0.x; a[9]  = (_Float16)b0.y; a[10] = (_Float16)b0.z; a[11] = (_Float16)b0.w;
  a[12] = (_Float16)b1.x; a[13] = (_Float16)b1.y; a[14] = (_Float16)b1.z; a[15] = (_Float16)b1.w;
  return a;
}

// Guarded A loader (K not multiple of 32). Branch-free: clamp address, select 0.
__device__ __forceinline__ v16h frag_a_guard(const float* __restrict__ A, int lda,
                                             int K, int m0, int k0, int lane) {
  int half = lane >> 4;
  int row  = m0 + (lane & 15);
  const float* p = A + (long)row * lda;
  v16h a;
#pragma unroll
  for (int e = 0; e < 16; ++e) {
    int k = k0 + half * 8 + ((e < 8) ? e : (e + 8));
    int kc = (k < K) ? k : 0;
    float v = p[kc];
    a[e] = (_Float16)((k < K) ? v : 0.0f);
  }
  return a;
}

__device__ __forceinline__ v8f wmma_f16(v16h a, v16h b, v8f c) {
  return __builtin_amdgcn_wmma_f32_16x16x32_f16(false, a, false, b, (short)0, c,
                                                false, false);
}

// ---------------- Generic WMMA GEMM:  C = act(alpha*(A@Wp) + bias) ---------
// A:  f32 [M,lda] (first K cols);  Wp: packed f16 fragments
// Wp layout: [(kt*ntiles + nt)*32 + lane][16]   (one v16h per lane per step)
// act: 0 none, 1 relu, 2 softplus.  acc: 1 -> C += result.
template <bool GUARDK>
__global__ void __launch_bounds__(256)
gemm_wmma_kernel(const float* __restrict__ A, const _Float16* __restrict__ Wp,
                 const float* __restrict__ bias, float* __restrict__ C,
                 int M, int N, int K, int lda, float alpha, int act, int acc) {
  int wave = threadIdx.x >> 5;
  int lane = threadIdx.x & 31;
  int half = lane >> 4;
  int m0 = blockIdx.y * 128 + wave * 16;
  int nt = blockIdx.x;                    // n-tile index
  if (m0 >= M) return;                    // wave-uniform
  int ntiles = N >> 4;
  int ktiles = (K + 31) >> 5;
  const v16h* Wv = (const v16h*)Wp;
  v8f c = {};
  for (int kt = 0; kt < ktiles; ++kt) {   // wave-uniform loop
    if (kt + 1 < ktiles)
      __builtin_prefetch(&Wv[((long)(kt + 1) * ntiles + nt) * 32], 0, 0);
    v16h a = GUARDK ? frag_a_guard(A, lda, K, m0, kt * 32, lane)
                    : frag_a_fast(A, lda, m0, kt * 32, lane);
    v16h b = Wv[((long)kt * ntiles + nt) * 32 + lane];
    c = wmma_f16(a, b, c);
  }
  int col = (nt << 4) + (lane & 15);      // N%16==0: always in range
  float bv = bias ? bias[col] : 0.0f;
#pragma unroll
  for (int r = 0; r < 8; ++r) {
    int row = m0 + r + 8 * half;          // M%16==0: always in range
    float v = c[r] * alpha + bv;
    if (act == 1)      v = fmaxf(v, 0.0f);
    else if (act == 2) v = (v > 20.0f) ? v : log1pf(expf(v));
    long o = (long)row * N + col;
    if (acc) C[o] += v; else C[o] = v;
  }
}

// ---------------- Weight packing -------------------------------------------
// Pack row-major f32 W[K,N] into fragment-major f16. Zero-pads k >= K.
__global__ void pack_frag_f32(const float* __restrict__ W, _Float16* __restrict__ P,
                              int K, int N, long total) {
  int ntiles = N >> 4;
  for (long idx = (long)blockIdx.x * blockDim.x + threadIdx.x; idx < total;
       idx += (long)gridDim.x * blockDim.x) {
    int  e    = (int)(idx & 15);
    long r    = idx >> 4;
    int  lane = (int)(r & 31);
    long tile = r >> 5;
    int  nt   = (int)(tile % ntiles);
    int  kt   = (int)(tile / ntiles);
    int  half = lane >> 4;
    int  k    = kt * 32 + half * 8 + ((e < 8) ? e : (e + 8));  // B layout: table order, matches A
    int  col  = nt * 16 + (lane & 15);
    P[idx] = (_Float16)((k < K) ? W[(long)k * N + col] : 0.0f);
  }
}

// remap agan_w_gat (4,128,32) -> f32 [128 (i), 128 (h*32+o)]
__global__ void remap_wgat_f32(const float* __restrict__ s, float* __restrict__ d) {
  int idx = blockIdx.x * blockDim.x + threadIdx.x;
  if (idx >= 128 * 128) return;
  int i = idx >> 7, ho = idx & 127, h = ho >> 5, o = ho & 31;
  d[idx] = s[(h * 128 + i) * 32 + o];
}
// remap pdm_w (4,256,64) -> f32 [256 (d), 256 (k*64+c)]
__global__ void remap_pdm_f32(const float* __restrict__ s, float* __restrict__ d) {
  int idx = blockIdx.x * blockDim.x + threadIdx.x;
  if (idx >= 256 * 256) return;
  int dd = idx >> 8, kc = idx & 255, k = kc >> 6, c = kc & 63;
  d[idx] = s[(k * 256 + dd) * 64 + c];
}
// sign_protos (2000,256) -> normalized, transposed f32 [256,2000]
__global__ void proto_normT_f32(const float* __restrict__ p, float* __restrict__ W,
                                int S, int D) {
  int s = blockIdx.x * blockDim.x + threadIdx.x;
  if (s >= S) return;
  float n = 0.0f;
  for (int d = 0; d < D; ++d) { float v = p[(long)s * D + d]; n += v * v; }
  float inv = 1.0f / (sqrtf(n) + 1e-8f);
  for (int d = 0; d < D; ++d) W[(long)d * S + s] = p[(long)s * D + d] * inv;
}

// ---------------- Fused GAT kernel: one block per (b,t) --------------------
// LDS A operand kept in fragment-major layout: index for (row, k):
//   mt=row/16, kt=k/32, kk=k%32, half=(kk>>3)&1, lane=(row%16)+16*half,
//   e = kk - half*8 - (kk<16 ? 0 : 8)
__device__ __forceinline__ int afrag_idx(int row, int k) {
  int mt = row >> 4, r = row & 15;
  int kt = k >> 5,  kk = k & 31;
  int half = (kk >> 3) & 1;
  int e = kk - half * 8 - ((kk < 16) ? 0 : 8);
  return (((mt * 4 + kt) * 32) + r + (half << 4)) * 16 + e;
}

__global__ void __launch_bounds__(256)
gat_fused_kernel(const float* __restrict__ x, const float* __restrict__ w_in,
                 const float* __restrict__ b_in, const _Float16* __restrict__ Wg,
                 const float* __restrict__ a_src, const float* __restrict__ a_dst,
                 const _Float16* __restrict__ Wout, const float* __restrict__ b_out,
                 float* __restrict__ spatial) {
  __shared__ __attribute__((aligned(32))) _Float16 sA[2 * 4 * 32 * 16]; // frag-major
  __shared__ float sH[32 * HIDD];
  __shared__ float sH0[NNODE * HIDD];
  __shared__ float sAl[NNODE * NNODE * 4];
  __shared__ float sSrc[NNODE * 4];
  __shared__ float sDst[NNODE * 4];
  __shared__ float sSpat[DOUTD];

  int bt = blockIdx.x;
  const float* xp = x + (long)bt * NNODE * 3;
  int tid = threadIdx.x;
  int wave = tid >> 5, lane = tid & 31, half = lane >> 4;
  const v16h* sAv = (const v16h*)sA;

  // 1) h0 = relu(x @ w_in + b_in) -> sA (f16 frag layout) + sH0 (f32)
  for (int idx = tid; idx < 32 * HIDD; idx += 256) {
    int n = idx >> 7, o = idx & 127;
    float v = 0.0f;
    if (n < NNODE) {
      v = b_in[o];
      for (int i = 0; i < 3; ++i) v += xp[n * 3 + i] * w_in[i * HIDD + o];
      v = fmaxf(v, 0.0f);
      sH0[n * HIDD + o] = v;
    }
    sA[afrag_idx(n, o)] = (_Float16)v;
  }
  __syncthreads();

  // 2) h = h0 @ Wg ([32,128]@[128,128]), 16 tiles / 8 waves (Wg ntiles=8)
  for (int t = wave; t < 16; t += 8) {
    int mt = t >> 3, nt = t & 7;
    v8f c = {};
    for (int kt = 0; kt < 4; ++kt) {
      v16h a = sAv[(mt * 4 + kt) * 32 + lane];
      v16h b = ((const v16h*)Wg)[(kt * 8 + nt) * 32 + lane];
      c = wmma_f16(a, b, c);
    }
    int col = nt * 16 + (lane & 15);
#pragma unroll
    for (int r = 0; r < 8; ++r) sH[(mt * 16 + r + 8 * half) * HIDD + col] = c[r];
  }
  __syncthreads();

  // 3) src/dst per (node, head)
  for (int idx = tid; idx < NNODE * 4; idx += 256) {
    int n = idx >> 2, hh = idx & 3;
    float s = 0.0f, d = 0.0f;
    for (int o = 0; o < 32; ++o) {
      float hv = sH[n * HIDD + hh * 32 + o];
      s += hv * a_src[hh * 32 + o];
      d += hv * a_dst[hh * 32 + o];
    }
    sSrc[idx] = s; sDst[idx] = d;
  }
  __syncthreads();

  // 4) alpha = softmax_j(leaky_relu(src_i + dst_j))
  for (int idx = tid; idx < NNODE * 4; idx += 256) {
    int i = idx >> 2, hh = idx & 3;
    float ev[NNODE];
    float mx = -1e30f;
    float si = sSrc[i * 4 + hh];
    for (int j = 0; j < NNODE; ++j) {
      float e = si + sDst[j * 4 + hh];
      e = (e > 0.0f) ? e : 0.2f * e;
      ev[j] = e; mx = fmaxf(mx, e);
    }
    float sum = 0.0f;
    for (int j = 0; j < NNODE; ++j) { ev[j] = expf(ev[j] - mx); sum += ev[j]; }
    float inv = 1.0f / sum;
    for (int j = 0; j < NNODE; ++j) sAl[(i * NNODE + j) * 4 + hh] = ev[j] * inv;
  }
  __syncthreads();

  // 5) feat = relu(h0 + alpha@h) -> sA (frag layout, padded rows zero)
  for (int idx = tid; idx < 32 * HIDD; idx += 256) {
    int n = idx >> 7, o = idx & 127;
    float v = 0.0f;
    if (n < NNODE) {
      int hh = o >> 5;
      float acc = 0.0f;
      for (int j = 0; j < NNODE; ++j)
        acc += sAl[(n * NNODE + j) * 4 + hh] * sH[j * HIDD + o];
      v = fmaxf(sH0[n * HIDD + o] + acc, 0.0f);
    }
    sA[afrag_idx(n, o)] = (_Float16)v;
  }
  sSpat[tid] = 0.0f;
  __syncthreads();

  // 6) featout = feat @ Wout ([32,128]@[128,256]); node-sum via ds_add_f32
  for (int t = wave; t < 32; t += 8) {
    int mt = t >> 4, nt = t & 15;
    v8f c = {};
    for (int kt = 0; kt < 4; ++kt) {
      v16h a = sAv[(mt * 4 + kt) * 32 + lane];
      v16h b = ((const v16h*)Wout)[(kt * 16 + nt) * 32 + lane];
      c = wmma_f16(a, b, c);
    }
    float part = 0.0f;
#pragma unroll
    for (int r = 0; r < 8; ++r) {
      int row = mt * 16 + r + 8 * half;
      if (row < NNODE) part += c[r];
    }
    atomicAdd(&sSpat[nt * 16 + (lane & 15)], part);
  }
  __syncthreads();
  spatial[(long)bt * DOUTD + tid] = sSpat[tid] * (1.0f / NNODE) + b_out[tid];
}

// ---------------- VALU kernels ---------------------------------------------
__global__ void __launch_bounds__(256)
ln_kernel(const float* __restrict__ x, const float* __restrict__ w, float* __restrict__ y) {
  __shared__ float red[256];
  long row = blockIdx.x;
  int tid = threadIdx.x;
  float v = x[row * DMODELD + tid];
  red[tid] = v; __syncthreads();
  for (int s = 128; s > 0; s >>= 1) { if (tid < s) red[tid] += red[tid + s]; __syncthreads(); }
  float mu = red[0] * (1.0f / DMODELD); __syncthreads();
  float dv = v - mu;
  red[tid] = dv * dv; __syncthreads();
  for (int s = 128; s > 0; s >>= 1) { if (tid < s) red[tid] += red[tid + s]; __syncthreads(); }
  float var = red[0] * (1.0f / DMODELD);
  y[row * DMODELD + tid] = dv * rsqrtf(var + 1e-5f) * w[tid];
}

__global__ void conv_silu_kernel(const float* __restrict__ xz, const float* __restrict__ w,
                                 const float* __restrict__ bc, float* __restrict__ u, int rev) {
  int idx = blockIdx.x * blockDim.x + threadIdx.x;
  if (idx >= BB * DINNERD) return;
  int b = idx / DINNERD, c = idx % DINNERD;
  float w0 = w[c * 4 + 0], w1 = w[c * 4 + 1], w2 = w[c * 4 + 2], w3 = w[c * 4 + 3];
  float bb = bc[c];
  float h1 = 0.0f, h2 = 0.0f, h3 = 0.0f;
  for (int t = 0; t < TT; ++t) {
    int s = rev ? (TT - 1 - t) : t;
    long row = (long)b * TT + s;
    float v = xz[row * 1024 + c];
    float y = w0 * h1 + w1 * h2 + w2 * h3 + w3 * v + bb;
    float sg = 1.0f / (1.0f + expf(-y));
    u[row * DINNERD + c] = y * sg;
    h1 = h2; h2 = h3; h3 = v;
  }
}

__global__ void scan_kernel(const float* __restrict__ dtf, const float* __restrict__ xdbl,
                            const float* __restrict__ xz, const float* __restrict__ A_log,
                            const float* __restrict__ Dp, float* __restrict__ uy, int rev) {
  int idx = blockIdx.x * blockDim.x + threadIdx.x;
  if (idx >= BB * DINNERD) return;
  int b = idx / DINNERD, c = idx % DINNERD;
  float Av[DSTATED];
#pragma unroll
  for (int j = 0; j < DSTATED; ++j) Av[j] = -expf(A_log[c * DSTATED + j]);
  float Dd = Dp[c];
  float h[DSTATED];
#pragma unroll
  for (int j = 0; j < DSTATED; ++j) h[j] = 0.0f;
  for (int t = 0; t < TT; ++t) {
    int s = rev ? (TT - 1 - t) : t;
    long row = (long)b * TT + s;
    float dt = dtf[row * DINNERD + c];
    float u  = uy[row * DINNERD + c];
    float y  = 0.0f;
#pragma unroll
    for (int j = 0; j < DSTATED; ++j) {
      float dA = expf(dt * Av[j]);
      h[j] = dA * h[j] + dt * xdbl[row * 48 + 16 + j] * u;
      y += h[j] * xdbl[row * 48 + 32 + j];
    }
    y += u * Dd;
    float z  = xz[row * 1024 + DINNERD + c];
    float sz = z / (1.0f + expf(-z));
    uy[row * DINNERD + c] = y * sz;
  }
}

__global__ void meant_kernel(const float* __restrict__ x, float* __restrict__ y, int D) {
  int idx = blockIdx.x * blockDim.x + threadIdx.x;
  if (idx >= BB * D) return;
  int b = idx / D, c = idx % D;
  float s = 0.0f;
  for (int t = 0; t < TT; ++t) s += x[((long)b * TT + t) * D + c];
  y[idx] = s * (1.0f / TT);
}

__global__ void rownorm_kernel(const float* __restrict__ x, float* __restrict__ y,
                               int R, int D) {
  int r = blockIdx.x * blockDim.x + threadIdx.x;
  if (r >= R) return;
  float s = 0.0f;
  for (int c = 0; c < D; ++c) { float v = x[(long)r * D + c]; s += v * v; }
  float inv = 1.0f / (sqrtf(s) + 1e-8f);
  for (int c = 0; c < D; ++c) y[(long)r * D + c] = x[(long)r * D + c] * inv;
}

__global__ void sims_kernel(const float* __restrict__ pooled, int koff,
                            const float* __restrict__ protos, int P, float* __restrict__ out) {
  int idx = blockIdx.x * blockDim.x + threadIdx.x;
  if (idx >= BB * P) return;
  int b = idx / P, p = idx % P;
  const float* f = pooled + b * 256 + koff;
  float nf = 0.0f, np = 0.0f, dot = 0.0f;
  for (int c = 0; c < 64; ++c) {
    float fv = f[c], pv = protos[p * 64 + c];
    nf += fv * fv; np += pv * pv; dot += fv * pv;
  }
  out[idx] = dot / ((sqrtf(nf) + 1e-8f) * (sqrtf(np) + 1e-8f)) * TEMP_INV;
}

// ---------------------------------------------------------------------------
static void launch_gemm(const float* A, const _Float16* Wp, const float* bias, float* C,
                        int M, int N, int K, int lda, float alpha, int act, int acc,
                        hipStream_t st) {
  dim3 g(N / 16, (M + 127) / 128);
  if (K % 32 == 0)
    gemm_wmma_kernel<false><<<g, 256, 0, st>>>(A, Wp, bias, C, M, N, K, lda, alpha, act, acc);
  else
    gemm_wmma_kernel<true><<<g, 256, 0, st>>>(A, Wp, bias, C, M, N, K, lda, alpha, act, acc);
}

extern "C" void kernel_launch(void* const* d_in, const int* in_sizes, int n_in,
                              void* d_out, int out_size, void* d_ws, size_t ws_size,
                              hipStream_t stream) {
  (void)in_sizes; (void)n_in; (void)out_size; (void)ws_size;
  const float* x        = (const float*)d_in[0];
  const float* w_in     = (const float*)d_in[1];
  const float* b_in     = (const float*)d_in[2];
  const float* w_gat    = (const float*)d_in[3];
  const float* a_src    = (const float*)d_in[4];
  const float* a_dst    = (const float*)d_in[5];
  const float* w_out    = (const float*)d_in[6];
  const float* b_out    = (const float*)d_in[7];
  const float* pdm_w    = (const float*)d_in[8];
  const float* pdm_b    = (const float*)d_in[9];
  const float* w_fuse   = (const float*)d_in[10];
  const float* b_fuse   = (const float*)d_in[11];
  const float* ln_w     = (const float*)d_in[12];
  const float* in_proj  = (const float*)d_in[13];
  const float* conv_w   = (const float*)d_in[14];
  const float* conv_b   = (const float*)d_in[15];
  const float* x_proj   = (const float*)d_in[16];
  const float* dt_w     = (const float*)d_in[17];
  const float* dt_b     = (const float*)d_in[18];
  const float* A_log    = (const float*)d_in[19];
  const float* Dp       = (const float*)d_in[20];
  const float* out_proj = (const float*)d_in[21];
  const float* w_sign   = (const float*)d_in[22];
  const float* b_sign   = (const float*)d_in[23];
  const float* protos   = (const float*)d_in[24];
  const float* p_hs     = (const float*)d_in[25];
  const float* p_loc    = (const float*)d_in[26];
  const float* p_mov    = (const float*)d_in[27];
  const float* p_ori    = (const float*)d_in[28];
  float* out = (float*)d_out;

  // ----- workspace layout -----
  char* ws = (char*)d_ws;
  size_t off = 0;
  auto alloc = [&](size_t b) { size_t o = off; off += (b + 255) & ~(size_t)255; return o; };
  _Float16* Wg    = (_Float16*)(ws + alloc((size_t)128 * 128 * 2));
  _Float16* Wout  = (_Float16*)(ws + alloc((size_t)128 * 256 * 2));
  _Float16* Wpdm  = (_Float16*)(ws + alloc((size_t)256 * 256 * 2));
  _Float16* Wfuse = (_Float16*)(ws + alloc((size_t)256 * 256 * 2));
  _Float16* Win   = (_Float16*)(ws + alloc((size_t)8 * 256 * 1024 * 2));
  _Float16* Wx    = (_Float16*)(ws + alloc((size_t)8 * 512 * 48 * 2));
  _Float16* Wdt   = (_Float16*)(ws + alloc((size_t)8 * 32 * 512 * 2));   // K padded to 32
  _Float16* Wop   = (_Float16*)(ws + alloc((size_t)8 * 512 * 256 * 2));
  _Float16* Whpc  = (_Float16*)(ws + alloc((size_t)256 * 256 * 2));
  _Float16* Pn    = (_Float16*)(ws + alloc((size_t)256 * 2000 * 2));
  float* tmpw    = (float*)(ws + alloc((size_t)256 * 2000 * 4));   // remap staging
  float* spatial = (float*)(ws + alloc((size_t)NBT * 256 * 4));
  float* comps   = (float*)(ws + alloc((size_t)NBT * 256 * 4));
  float* xx      = (float*)(ws + alloc((size_t)NBT * 256 * 4));
  float* xn      = (float*)(ws + alloc((size_t)NBT * 256 * 4));
  float* xzb     = (float*)(ws + alloc((size_t)NBT * 1024 * 4));
  float* uconv   = (float*)(ws + alloc((size_t)NBT * 512 * 4));
  float* xdbl    = (float*)(ws + alloc((size_t)NBT * 48 * 4));
  float* dtf     = (float*)(ws + alloc((size_t)NBT * 512 * 4));
  float* tm      = (float*)(ws + alloc((size_t)BB * 256 * 4));
  float* fnorm   = (float*)(ws + alloc((size_t)BB * 256 * 4));
  float* pooled  = (float*)(ws + alloc((size_t)BB * 256 * 4));

  auto pack = [&](const float* src, _Float16* dst, int K, int N) {
    long tot = (long)((K + 31) >> 5) * (N >> 4) * 512;
    int blocks = (int)((tot + 255) / 256);
    if (blocks > 2048) blocks = 2048;
    pack_frag_f32<<<blocks, 256, 0, stream>>>(src, dst, K, N, tot);
  };

  // ----- weight packing -----
  remap_wgat_f32<<<(128 * 128 + 255) / 256, 256, 0, stream>>>(w_gat, tmpw);
  pack(tmpw, Wg, 128, 128);
  pack(w_out, Wout, 128, 256);
  remap_pdm_f32<<<(256 * 256 + 255) / 256, 256, 0, stream>>>(pdm_w, tmpw);
  pack(tmpw, Wpdm, 256, 256);
  pack(w_fuse, Wfuse, 256, 256);
  for (int ld = 0; ld < 8; ++ld) {
    pack(in_proj  + (size_t)ld * 256 * 1024, Win + (size_t)ld * 256 * 1024, 256, 1024);
    pack(x_proj   + (size_t)ld * 512 * 48,   Wx  + (size_t)ld * 512 * 48,   512, 48);
    pack(dt_w     + (size_t)ld * 16 * 512,   Wdt + (size_t)ld * 32 * 512,   16,  512);
    pack(out_proj + (size_t)ld * 512 * 256,  Wop + (size_t)ld * 512 * 256,  512, 256);
  }
  pack(w_sign, Whpc, 256, 256);
  proto_normT_f32<<<(2000 + 255) / 256, 256, 0, stream>>>(protos, tmpw, 2000, 256);
  pack(tmpw, Pn, 256, 2000);

  // ----- GAT + spatial pooling -----
  gat_fused_kernel<<<NBT, 256, 0, stream>>>(x, w_in, b_in, Wg, a_src, a_dst,
                                            Wout, b_out, spatial);

  // ----- comps / fused -----
  launch_gemm(spatial, Wpdm, pdm_b, comps, NBT, 256, 256, 256, 1.0f, 1, 0, stream);
  launch_gemm(comps, Wfuse, b_fuse, xx, NBT, 256, 256, 256, 1.0f, 0, 0, stream);

  // ----- 4 bidirectional SSM layers -----
  for (int l = 0; l < 4; ++l) {
    ln_kernel<<<NBT, 256, 0, stream>>>(xx, ln_w + l * 256, xn);
    for (int d = 0; d < 2; ++d) {
      int ld = l * 2 + d;
      int rev = d;
      launch_gemm(xn, Win + (size_t)ld * 256 * 1024, nullptr, xzb,
                  NBT, 1024, 256, 256, 1.0f, 0, 0, stream);
      conv_silu_kernel<<<(BB * DINNERD + 255) / 256, 256, 0, stream>>>(
          xzb, conv_w + (size_t)ld * 512 * 4, conv_b + (size_t)ld * 512, uconv, rev);
      launch_gemm(uconv, Wx + (size_t)ld * 512 * 48, nullptr, xdbl,
                  NBT, 48, 512, 512, 1.0f, 0, 0, stream);
      launch_gemm(xdbl, Wdt + (size_t)ld * 32 * 512, dt_b + (size_t)ld * 512, dtf,
                  NBT, 512, 16, 48, 1.0f, 2, 0, stream);   // K=16 guarded
      scan_kernel<<<(BB * DINNERD + 255) / 256, 256, 0, stream>>>(
          dtf, xdbl, xzb, A_log + (size_t)ld * 512 * 16, Dp + (size_t)ld * 512,
          uconv, rev);
      launch_gemm(uconv, Wop + (size_t)ld * 512 * 256, nullptr, xx,
                  NBT, 256, 512, 512, 1.0f, 0, 1, stream);
    }
  }

  // ----- heads -----
  meant_kernel<<<(BB * 256 + 255) / 256, 256, 0, stream>>>(xx, tm, 256);
  meant_kernel<<<(BB * 256 + 255) / 256, 256, 0, stream>>>(comps, pooled, 256);

  float* emb = out + 128000;
  launch_gemm(tm, Whpc, b_sign, emb, BB, 256, 256, 256, 1.0f, 0, 0, stream);
  rownorm_kernel<<<1, 64, 0, stream>>>(emb, fnorm, BB, 256);
  launch_gemm(fnorm, Pn, nullptr, out, BB, 2000, 256, 256, TEMP_INV, 0, 0, stream);

  float* o_hs  = out + 128000 + 16384;
  float* o_loc = o_hs + 64 * 40;
  float* o_mov = o_loc + 64 * 20;
  float* o_ori = o_mov + 64 * 20;
  sims_kernel<<<(BB * 40 + 255) / 256, 256, 0, stream>>>(pooled, 0 * 64, p_hs, 40, o_hs);
  sims_kernel<<<(BB * 20 + 255) / 256, 256, 0, stream>>>(pooled, 1 * 64, p_loc, 20, o_loc);
  sims_kernel<<<(BB * 20 + 255) / 256, 256, 0, stream>>>(pooled, 2 * 64, p_mov, 20, o_mov);
  sims_kernel<<<(BB * 8 + 255) / 256, 256, 0, stream>>>(pooled, 3 * 64, p_ori, 8, o_ori);
}
